// CausalSelfAttention_46273977647772
// MI455X (gfx1250) — matrix operand order repaired and checked by reading, not compiled, not run on hardware
//
#include <hip/hip_runtime.h>

typedef __attribute__((ext_vector_type(16))) _Float16 v16h;
typedef __attribute__((ext_vector_type(8)))  _Float16 v8h;
typedef __attribute__((ext_vector_type(8)))  float    v8f;
typedef int v4i __attribute__((vector_size(16)));

#define B_   4
#define T_   2048
#define C_   2048
#define H_   16
#define D_   128
#define MTOK (B_ * T_)   // 8192

#if defined(__gfx1250__) && \
    __has_builtin(__builtin_amdgcn_global_load_async_to_lds_b128) && \
    __has_builtin(__builtin_amdgcn_s_wait_asynccnt)
#define USE_ASYNC 1
#else
#define USE_ASYNC 0
#endif

union Frag16 { v16h v; v8h h8[2]; };

__device__ __forceinline__ v8f wmma16x16x32(const Frag16& a, const Frag16& b, v8f c) {
    // D = A(16x32 f16) x B(32x16 f16) + C(16x16 f32)
    return __builtin_amdgcn_wmma_f32_16x16x32_f16(false, a.v, false, b.v,
                                                  (short)0, c, false, false);
}

// 16-byte global -> LDS copy: async (GLOBAL_LOAD_ASYNC_TO_LDS_B128) when available
__device__ __forceinline__ void copy16(const _Float16* g, _Float16* l) {
#if USE_ASYNC
    __builtin_amdgcn_global_load_async_to_lds_b128(
        (__attribute__((address_space(1))) v4i*)g,
        (__attribute__((address_space(3))) v4i*)l, 0, 0);
#else
    *(v8h*)l = *(const v8h*)g;
#endif
}

__device__ __forceinline__ void wait_async_all() {
#if USE_ASYNC
    __builtin_amdgcn_s_wait_asynccnt(0);
#endif
}

// ---------------------------------------------------------------------------
// fp32 -> f16 conversion (vectorized: float4 in, 4x f16 out)
// ---------------------------------------------------------------------------
__global__ void cvt_f32_to_f16(const float* __restrict__ x, _Float16* __restrict__ xh, int n4) {
    int i = blockIdx.x * 256 + threadIdx.x;
    if (i >= n4) return;
    float4 v = reinterpret_cast<const float4*>(x)[i];
    union { _Float16 h[4]; uint2 u; } p;
    p.h[0] = (_Float16)v.x; p.h[1] = (_Float16)v.y;
    p.h[2] = (_Float16)v.z; p.h[3] = (_Float16)v.w;
    reinterpret_cast<uint2*>(xh)[i] = p.u;
}

// ---------------------------------------------------------------------------
// W (n,k) fp32 -> Wt (k,n) f16, 32x32 LDS tile transpose
// ---------------------------------------------------------------------------
__global__ void transpose_cvt(const float* __restrict__ W, _Float16* __restrict__ Wt) {
    __shared__ float tile[32][33];
    const int n0 = blockIdx.x * 32, k0 = blockIdx.y * 32;
    const int tx = threadIdx.x, ty = threadIdx.y;
    for (int i = 0; i < 32; i += 8)
        tile[ty + i][tx] = W[(size_t)(n0 + ty + i) * C_ + k0 + tx];
    __syncthreads();
    for (int i = 0; i < 32; i += 8)
        Wt[(size_t)(k0 + ty + i) * C_ + n0 + tx] = (_Float16)tile[tx][ty + i];
}

// ---------------------------------------------------------------------------
// f16 GEMM: out(m,n) = A(m,:) . Bm(:,n) + bias(n)
//   A:  row-major (MTOK x C_) f16        Bm: row-major (C_ x C_) f16 (= W^T)
//   MODE 0: fp32 out row-major (final projection)
//   MODE 1: f16 out as (B,H,T,D)  (Q and V)
//   MODE 2: f16 out as (B,H,D,T)  (K pre-transposed for QK^T)
// Block: 128x128 tile, 256 threads (8 waves), wave tile 64x32 (4x2 WMMA tiles)
// Double-buffered LDS, async global->LDS staging.
// ---------------------------------------------------------------------------
template <int MODE>
__global__ __launch_bounds__(256) void gemm_f16(const _Float16* __restrict__ A,
                                                const _Float16* __restrict__ Bm,
                                                const float* __restrict__ bias,
                                                void* __restrict__ outp) {
    __shared__ alignas(16) _Float16 As[2][128 * 40];  // 32 K-cols, stride padded to 40
    __shared__ alignas(16) _Float16 Bs[2][32 * 136];  // 128 N-cols, stride padded to 136

    const int tid  = threadIdx.x;
    const int lane = tid & 31, wid = tid >> 5;
    const int wm = wid >> 2, wn = wid & 3;            // 2x4 wave grid
    const int mbase = blockIdx.y * 128;
    const int nbase = blockIdx.x * 128;

    v8f acc[4][2];
    for (int mt = 0; mt < 4; ++mt)
        for (int nt = 0; nt < 2; ++nt)
            acc[mt][nt] = (v8f){0.f,0.f,0.f,0.f,0.f,0.f,0.f,0.f};

    const int kk = (lane & 15) + ((lane >> 4) << 4);  // B-frag K row: 0..31
    const int kq = (lane >> 4) * 8;                   // A-frag K sub-offset

    // stage one 128x32 A tile + 32x128 B tile (16B chunks, 4 per thread)
    auto stage = [&](int k0, int buf) {
        for (int c = tid; c < 512; c += 256) {        // As: 128 rows x 32 f16
            int r = c >> 2, j = c & 3;
            copy16(A + (size_t)(mbase + r) * C_ + k0 + j * 8, As[buf] + r * 40 + j * 8);
        }
        for (int c = tid; c < 512; c += 256) {        // Bs: 32 rows x 128 f16
            int r = c >> 4, j = c & 15;
            copy16(Bm + (size_t)(k0 + r) * C_ + nbase + j * 8, Bs[buf] + r * 136 + j * 8);
        }
    };

    stage(0, 0);
    for (int it = 0; it < C_ / 32; ++it) {
        const int cur = it & 1;
        wait_async_all();          // current buffer landed in LDS
        __syncthreads();           // ... for every wave; prev buffer fully consumed
        if (it + 1 < C_ / 32)
            stage((it + 1) * 32, cur ^ 1);   // overlaps with WMMA below

        Frag16 bf[2];
        for (int nt = 0; nt < 2; ++nt) {
            const int col = wn * 32 + nt * 16;
            for (int s = 0; s < 8; ++s) {
                bf[nt].v[s]     = Bs[cur][(kq + s) * 136 + col + (lane & 15)];
                bf[nt].v[s + 8] = Bs[cur][(16 + kq + s) * 136 + col + (lane & 15)];
            }
        }
        for (int mt = 0; mt < 4; ++mt) {
            Frag16 af;
            const int ra = wm * 64 + mt * 16 + (lane & 15);
            af.h8[0] = *(const v8h*)(As[cur] + ra * 40 + kq);
            af.h8[1] = *(const v8h*)(As[cur] + ra * 40 + 16 + kq);
            for (int nt = 0; nt < 2; ++nt)
                acc[mt][nt] = wmma16x16x32(af, bf[nt], acc[mt][nt]);
        }
    }

    // epilogue: lane holds (m = i + 8*(lane>>4), n = lane&15) per v8f
    for (int mt = 0; mt < 4; ++mt)
        for (int nt = 0; nt < 2; ++nt) {
            const int n  = nbase + wn * 32 + nt * 16 + (lane & 15);
            const float bv = bias[n];
            for (int i = 0; i < 8; ++i) {
                const int m = mbase + wm * 64 + mt * 16 + i + 8 * (lane >> 4);
                const float val = acc[mt][nt][i] + bv;
                if (MODE == 0) {
                    ((float*)outp)[(size_t)m * C_ + n] = val;
                } else {
                    const int b = m >> 11, t = m & (T_ - 1);
                    const int h = n >> 7, d = n & (D_ - 1);
                    size_t idx;
                    if (MODE == 1) idx = (((size_t)(b * H_ + h)) * T_ + t) * D_ + d;
                    else           idx = (((size_t)(b * H_ + h)) * D_ + d) * T_ + t;
                    ((_Float16*)outp)[idx] = (_Float16)val;
                }
            }
        }
}

// ---------------------------------------------------------------------------
// Fused causal flash attention.
//   Q,V: (B,H,T,D) f16   Kt: (B,H,D,T) f16   am: (B,1,1,T) fp32
//   Y:   (B*T, C) f16 row-major (feeds final projection GEMM)
// Block = (b,h,64-row q tile), 128 threads = 4 waves, wave owns 16 q rows.
// Double-buffered async K/V staging.
// ---------------------------------------------------------------------------
__global__ __launch_bounds__(128) void attn_fused(const _Float16* __restrict__ Q,
                                                  const _Float16* __restrict__ Kt,
                                                  const _Float16* __restrict__ V,
                                                  const float* __restrict__ am,
                                                  _Float16* __restrict__ Y) {
    __shared__ alignas(16) _Float16 Ks[2][128 * 72]; // d-major, 64 keys (pad 72)
    __shared__ alignas(16) _Float16 Vs[2][64 * 136]; // key-major, 128 d (pad 136)
    __shared__ alignas(16) float    Sw[4][16 * 72];  // per-wave S scratch
    __shared__ alignas(16) _Float16 Ps[4][16 * 72];  // per-wave P (A-frag source)
    __shared__ float rowscale[4][16];
    __shared__ float rowsum[4][16];

    const int tid  = threadIdx.x;
    const int lane = tid & 31, wid = tid >> 5;
    const int qi = blockIdx.x, h = blockIdx.y, b = blockIdx.z;
    const int qbase = qi * 64;
    const size_t bh = (size_t)(b * H_ + h);
    const float SCALE = 0.08838834764831845f;   // 1/sqrt(128)
    const float NEG_BIG = -3.0e38f;

    // stage 64-key K^T tile (128 d x 64) and V tile (64 x 128 d)
    auto stage_kv = [&](int kb, int buf) {
        for (int c = tid; c < 1024; c += 128) {
            int d = c >> 3, j = c & 7;
            copy16(Kt + (bh * D_ + d) * T_ + kb + j * 8, Ks[buf] + d * 72 + j * 8);
        }
        for (int c = tid; c < 1024; c += 128) {
            int r = c >> 4, j = c & 15;
            copy16(V + (bh * T_ + kb + r) * D_ + j * 8, Vs[buf] + r * 136 + j * 8);
        }
    };

    // resident Q fragments: 16 q-rows x 128 d = 4 x (16x32) A-fragments
    Frag16 qa[4];
    {
        const int qg = qbase + wid * 16 + (lane & 15);
        const _Float16* qp = Q + (bh * T_ + qg) * D_;
        const int kq0 = (lane >> 4) * 8;
        for (int kf = 0; kf < 4; ++kf) {
            qa[kf].h8[0] = *(const v8h*)(qp + kf * 32 + kq0);
            qa[kf].h8[1] = *(const v8h*)(qp + kf * 32 + 16 + kq0);
        }
    }

    v8f O[8];
    for (int nt = 0; nt < 8; ++nt) O[nt] = (v8f){0.f,0.f,0.f,0.f,0.f,0.f,0.f,0.f};
    float mreg = NEG_BIG, lreg = 0.0f;

    const int kkb = (lane & 15) + ((lane >> 4) << 4);
    const int kq  = (lane >> 4) * 8;

    stage_kv(0, 0);
    for (int kj = 0; kj <= qi; ++kj) {         // causal: skip tiles past diagonal
        const int cur = kj & 1;
        wait_async_all();
        __syncthreads();                        // cur tile ready; prev tile consumed
        if (kj + 1 <= qi)
            stage_kv((kj + 1) * 64, cur ^ 1);   // overlaps with WMMA below
        const int kb = kj * 64;

        // S = Q K^T (4 key sub-tiles x 4 d-chunks of WMMA)
        for (int nt = 0; nt < 4; ++nt) {
            v8f a = (v8f){0.f,0.f,0.f,0.f,0.f,0.f,0.f,0.f};
            for (int kf = 0; kf < 4; ++kf) {
                Frag16 bf;
                const int krow = kkb + kf * 32;
                for (int s = 0; s < 8; ++s) {
                    bf.v[s]     = Ks[cur][(kf * 32 + kq + s) * 72 + nt * 16 + (lane & 15)];
                    bf.v[s + 8] = Ks[cur][(kf * 32 + 16 + kq + s) * 72 + nt * 16 + (lane & 15)];
                }
                a = wmma16x16x32(qa[kf], bf, a);
            }
            const int kg = kb + nt * 16 + (lane & 15);
            const float amv = am[(size_t)b * T_ + kg];
            for (int i = 0; i < 8; ++i) {
                const int m  = i + 8 * (lane >> 4);
                const int qg = qbase + wid * 16 + m;
                float s = a[i] * SCALE;
                s = (kg <= qg) ? (s + amv) : NEG_BIG;   // causal + additive mask
                Sw[wid][m * 72 + nt * 16 + (lane & 15)] = s;
            }
        }
        __syncthreads();

        // online softmax: lanes 0..15 each own one q row of this wave
        if (lane < 16) {
            const int m = lane;
            float tmax = NEG_BIG;
            for (int j = 0; j < 64; ++j) tmax = fmaxf(tmax, Sw[wid][m * 72 + j]);
            const float mnew = fmaxf(mreg, tmax);
            const float sc = __expf(mreg - mnew);
            float tsum = 0.f;
            for (int j = 0; j < 64; ++j) {
                const float p = __expf(Sw[wid][m * 72 + j] - mnew);
                tsum += p;
                Ps[wid][m * 72 + j] = (_Float16)p;
            }
            lreg = lreg * sc + tsum;
            mreg = mnew;
            rowscale[wid][m] = sc;
        }
        __syncthreads();

        // rescale running O, then O += P V
        {
            float f[8];
            for (int i = 0; i < 8; ++i) f[i] = rowscale[wid][i + 8 * (lane >> 4)];
            for (int nt = 0; nt < 8; ++nt)
                for (int i = 0; i < 8; ++i) O[nt][i] *= f[i];
        }
        for (int kf2 = 0; kf2 < 2; ++kf2) {
            Frag16 pf;
            const int r = lane & 15;
            pf.h8[0] = *(const v8h*)(Ps[wid] + r * 72 + kf2 * 32 + kq);
            pf.h8[1] = *(const v8h*)(Ps[wid] + r * 72 + kf2 * 32 + 16 + kq);
            const int krow = kkb + kf2 * 32;
            for (int nt = 0; nt < 8; ++nt) {
                Frag16 vf;
                for (int s = 0; s < 8; ++s) {
                    vf.v[s]     = Vs[cur][(kf2 * 32 + kq + s) * 136 + nt * 16 + (lane & 15)];
                    vf.v[s + 8] = Vs[cur][(kf2 * 32 + 16 + kq + s) * 136 + nt * 16 + (lane & 15)];
                }
                O[nt] = wmma16x16x32(pf, vf, O[nt]);
            }
        }
        // no trailing barrier: next iteration's top barrier guards buffer reuse
    }

    if (lane < 16) rowsum[wid][lane] = lreg;
    __syncthreads();

    {
        float inv[8];
        for (int i = 0; i < 8; ++i) inv[i] = 1.0f / rowsum[wid][i + 8 * (lane >> 4)];
        for (int nt = 0; nt < 8; ++nt) {
            const int n = h * D_ + nt * 16 + (lane & 15);
            for (int i = 0; i < 8; ++i) {
                const int m = qbase + wid * 16 + i + 8 * (lane >> 4);
                Y[((size_t)b * T_ + m) * C_ + n] = (_Float16)(O[nt][i] * inv[i]);
            }
        }
    }
}

// ---------------------------------------------------------------------------
extern "C" void kernel_launch(void* const* d_in, const int* in_sizes, int n_in,
                              void* d_out, int out_size, void* d_ws, size_t ws_size,
                              hipStream_t stream) {
    (void)in_sizes; (void)n_in; (void)out_size; (void)ws_size;
    const float* x  = (const float*)d_in[0];
    const float* am = (const float*)d_in[1];
    const float* Wq = (const float*)d_in[2];
    const float* bq = (const float*)d_in[3];
    const float* Wk = (const float*)d_in[4];
    const float* bk = (const float*)d_in[5];
    const float* Wv = (const float*)d_in[6];
    const float* bv = (const float*)d_in[7];
    const float* Wp = (const float*)d_in[8];
    const float* bp = (const float*)d_in[9];
    float* out = (float*)d_out;

    char* ws = (char*)d_ws;
    const size_t MB = 1ull << 20;
    _Float16* xh  = (_Float16*)(ws);             // 32 MB  x as f16
    _Float16* wqt = (_Float16*)(ws + 32 * MB);   //  8 MB  Wq^T f16
    _Float16* wkt = (_Float16*)(ws + 40 * MB);
    _Float16* wvt = (_Float16*)(ws + 48 * MB);
    _Float16* wpt = (_Float16*)(ws + 56 * MB);
    _Float16* Qh  = (_Float16*)(ws + 64 * MB);   // 32 MB  (B,H,T,D)
    _Float16* Kth = (_Float16*)(ws + 96 * MB);   // 32 MB  (B,H,D,T)
    _Float16* Vh  = (_Float16*)(ws + 128 * MB);  // 32 MB  (B,H,T,D)
    _Float16* Yh  = (_Float16*)(ws + 160 * MB);  // 32 MB  attention out f16

    // precision conversion / weight transpose
    cvt_f32_to_f16<<<(MTOK * C_ / 4 + 255) / 256, 256, 0, stream>>>(x, xh, MTOK * C_ / 4);
    dim3 tg(64, 64), tb(32, 8);
    transpose_cvt<<<tg, tb, 0, stream>>>(Wq, wqt);
    transpose_cvt<<<tg, tb, 0, stream>>>(Wk, wkt);
    transpose_cvt<<<tg, tb, 0, stream>>>(Wv, wvt);
    transpose_cvt<<<tg, tb, 0, stream>>>(Wp, wpt);

    // QKV projections
    dim3 gg(C_ / 128, MTOK / 128);   // (16, 64)
    gemm_f16<1><<<gg, 256, 0, stream>>>(xh, wqt, bq, (void*)Qh);
    gemm_f16<2><<<gg, 256, 0, stream>>>(xh, wkt, bk, (void*)Kth);
    gemm_f16<1><<<gg, 256, 0, stream>>>(xh, wvt, bv, (void*)Vh);

    // fused causal attention
    dim3 ga(T_ / 64, H_, B_);
    attn_fused<<<ga, 128, 0, stream>>>(Qh, Kth, Vh, am, Yh);

    // output projection -> fp32
    gemm_f16<0><<<gg, 256, 0, stream>>>(Yh, wpt, bp, (void*)out);
}
